// model_64321430225450
// MI455X (gfx1250) — compile-verified, matrix operand order checked by reading, not run on hardware
//
#include <hip/hip_runtime.h>

typedef __attribute__((ext_vector_type(16))) _Float16 v16h;
typedef __attribute__((ext_vector_type(8)))  _Float16 v8h;
typedef __attribute__((ext_vector_type(8)))  float    v8f;

#define DEVINL __device__ __forceinline__

DEVINL float elu_f(float x) { return x > 0.f ? x : __expf(x) - 1.f; }

// Load one 16x32 (f16) A/B fragment row for wave32 WMMA.
// Layout per ISA 7.12.2: lanes 0-15 hold K=[0..7]+[16..23], lanes 16-31 K=[8..15]+[24..31].
DEVINL v16h frag_load(const _Float16* row, int kbase, int hf) {
    v8h x0 = *(const v8h*)(row + kbase + hf * 8);
    v8h x1 = *(const v8h*)(row + kbase + 16 + hf * 8);
    return __builtin_shufflevector(x0, x1, 0,1,2,3,4,5,6,7,8,9,10,11,12,13,14,15);
}

DEVINL v8f wmma_f16(v16h a, v16h b, v8f c) {
    return __builtin_amdgcn_wmma_f32_16x16x32_f16(false, a, false, b, (short)0, c, false, false);
}

// ---- CDNA5 async global->LDS (ASYNCcnt-tracked), inline asm (no builtin probed) ----
DEVINL unsigned lds_addr32(const void* p) {            // generic addr low 32 bits == LDS offset
    return (unsigned)(uintptr_t)p;
}
DEVINL void async_ld_b128(unsigned lds, const _Float16* g) {
    asm volatile("global_load_async_to_lds_b128 %0, %1, off" :: "v"(lds), "v"(g) : "memory");
}
DEVINL void wait_async_le(int keep2) {                 // keep2: 1 -> wait <=2, 0 -> wait ==0
    if (keep2) asm volatile("s_wait_asynccnt 0x2" ::: "memory");
    else       asm volatile("s_wait_asynccnt 0x0" ::: "memory");
}

// ---------------------------------------------------------------- utilities
__global__ void cvt_f32_f16(const float* __restrict__ src, _Float16* __restrict__ dst, int n) {
    int i = blockIdx.x * blockDim.x + threadIdx.x;
    if (i < n) dst[i] = (_Float16)src[i];
}

// ---------------------------------------------------------------- embedding + reparam
// one block of 128 threads per batch element; thread = embedding dim e
__global__ void embed_kernel(const int* __restrict__ user, const int* __restrict__ item,
                             const float* __restrict__ eps_u, const float* __restrict__ eps_i,
                             const float* __restrict__ Wum, const float* __restrict__ bum,
                             const float* __restrict__ Wuv, const float* __restrict__ buv,
                             const float* __restrict__ Wim, const float* __restrict__ bim,
                             const float* __restrict__ Wiv, const float* __restrict__ biv,
                             _Float16* __restrict__ su, _Float16* __restrict__ si) {
    int b = blockIdx.x;
    int e = threadIdx.x;                 // 0..127
    size_t uo = (size_t)user[b] * 128 + e;
    size_t io = (size_t)item[b] * 128 + e;
    float um = Wum[uo] + bum[e];
    float us = elu_f(Wuv[uo] + buv[e]) + 1.f;
    float im = Wim[io] + bim[e];
    float is = elu_f(Wiv[io] + biv[e]) + 1.f;
    size_t base = (size_t)b * 9 * 128 + e;
    for (int k = 0; k < 9; ++k) {
        su[base + k * 128] = (_Float16)(eps_u[base + k * 128] * us + um);
        si[base + k * 128] = (_Float16)(eps_i[base + k * 128] * is + im);
    }
}

// ---------------------------------------------------------------- conv1 (1x1) as GEMM
// per item: D[81pos x 64ch] = fmap[81 x 128] x W1^T[128 x 64]; M padded to 96.
__global__ __launch_bounds__(128) void conv1_kernel(const _Float16* __restrict__ su,
                                                    const _Float16* __restrict__ si,
                                                    const _Float16* __restrict__ w1,  // [64][128]
                                                    const float* __restrict__ b1,
                                                    _Float16* __restrict__ act1) {    // [B][64][81]
    __shared__ _Float16 A[96 * 128];
    int b = blockIdx.x, tid = threadIdx.x;
    const _Float16* sub = su + (size_t)b * 9 * 128;
    const _Float16* sib = si + (size_t)b * 9 * 128;
    int e = tid;  // 0..127
    for (int p = 0; p < 96; ++p) {
        _Float16 v = (_Float16)0.f;
        if (p < 81) {
            int i = p / 9, j = p - i * 9;
            v = sub[i * 128 + e] * sib[j * 128 + e];
        }
        A[p * 128 + e] = v;
    }
    __syncthreads();

    int wv = tid >> 5, lane = tid & 31, l15 = lane & 15, hf = lane >> 4;
    int n = wv * 16 + l15;
    const _Float16* brow = w1 + (size_t)n * 128;
    float bias = b1[n];
    for (int mt = 0; mt < 6; ++mt) {
        v8f acc = {};
        const _Float16* arow = A + (mt * 16 + l15) * 128;
        for (int kt = 0; kt < 4; ++kt) {
            v16h af = frag_load(arow, kt * 32, hf);
            v16h bf = frag_load(brow, kt * 32, hf);
            acc = wmma_f16(af, bf, acc);
        }
        for (int v = 0; v < 8; ++v) {
            int p = mt * 16 + v + 8 * hf;
            if (p < 81)
                act1[(size_t)b * 64 * 81 + (size_t)n * 81 + p] = (_Float16)elu_f(acc[v] + bias);
        }
    }
}

// ---------------------------------------------------------------- conv2 (3x3) as im2col GEMM
// per item: D[49pos x 64ch] = im2col[49 x 576] x W2^T[576 x 64]; M padded to 64.
// activation preload uses async global->LDS (648 x 16B chunks).
__global__ __launch_bounds__(128) void conv2_kernel(const _Float16* __restrict__ act1, // [B][64][81]
                                                    const _Float16* __restrict__ w2,   // [64][576]
                                                    const float* __restrict__ b2,
                                                    _Float16* __restrict__ act2) {     // [B][3136]
    __shared__ _Float16 ach[64 * 81];
    __shared__ _Float16 At[64 * 32];
    int b = blockIdx.x, tid = threadIdx.x;
    const _Float16* src = act1 + (size_t)b * 64 * 81;
    {
        unsigned base = lds_addr32(ach);
        for (int idx = tid; idx < 648; idx += 128)          // 648 * 16B = 64*81*2
            async_ld_b128(base + idx * 16, src + idx * 8);
        wait_async_le(0);
    }
    __syncthreads();

    int wv = tid >> 5, lane = tid & 31, l15 = lane & 15, hf = lane >> 4;
    int n = wv * 16 + l15;
    const _Float16* brow = w2 + (size_t)n * 576;
    v8f acc[4] = {{}, {}, {}, {}};

    for (int kt = 0; kt < 18; ++kt) {
        int kbase = kt * 32;
        for (int idx = tid; idx < 64 * 32; idx += 128) {
            int p = idx >> 5, kk = idx & 31;
            int k = kbase + kk;
            int c = k / 9, r = k - c * 9;
            int kh = r / 3, kw = r - kh * 3;
            _Float16 v = (_Float16)0.f;
            if (p < 49) {
                int h = p / 7 + kh, w = (p - (p / 7) * 7) + kw;
                v = ach[c * 81 + h * 9 + w];
            }
            At[idx] = v;
        }
        __syncthreads();
        v16h bf = frag_load(brow, kbase, hf);
        for (int mt = 0; mt < 4; ++mt) {
            v16h af = frag_load(At + (mt * 16 + l15) * 32, 0, hf);
            acc[mt] = wmma_f16(af, bf, acc[mt]);
        }
        __syncthreads();
    }

    float bias = b2[n];
    for (int mt = 0; mt < 4; ++mt)
        for (int v = 0; v < 8; ++v) {
            int p = mt * 16 + v + 8 * hf;
            if (p < 49)
                act2[(size_t)b * 3136 + (size_t)n * 49 + p] = (_Float16)elu_f(acc[mt][v] + bias);
        }
}

// ---------------------------------------------------------------- fused FC GEMM + bias + BN + ELU
// C[M x N] = A[M x K] * B^T, B stored [N][K] f16. Block = 128 thr (4 waves), tile 64M x 64N.
// A tiles double-buffered in LDS via async global->LDS copies (2 B128 issues/thread/tile).
__global__ __launch_bounds__(128) void fc_gemm_kernel(const _Float16* __restrict__ A,
                                                      const _Float16* __restrict__ Bw,
                                                      const float* __restrict__ bias,
                                                      const float* __restrict__ g,
                                                      const float* __restrict__ bb,
                                                      const float* __restrict__ rm,
                                                      const float* __restrict__ rv,
                                                      _Float16* __restrict__ out,
                                                      int N, int Kdim) {
    __shared__ _Float16 At[2][64 * 32];
    int m0 = blockIdx.x * 64;
    int nb = blockIdx.y;
    int tid = threadIdx.x;
    int wv = tid >> 5, lane = tid & 31, l15 = lane & 15, hf = lane >> 4;
    int n = nb * 64 + wv * 16 + l15;
    const _Float16* brow = Bw + (size_t)n * Kdim;
    v8f acc[4] = {{}, {}, {}, {}};

    unsigned lbase[2] = { lds_addr32(At[0]), lds_addr32(At[1]) };

    // async-stage one 64x32 tile: 256 chunks of 16B; LDS byte offset == chunk*16
    auto stage = [&](int buf, int kb) {
        for (int c = tid; c < 256; c += 128) {
            int p = c >> 2, kk = (c & 3) * 8;
            async_ld_b128(lbase[buf] + c * 16, A + (size_t)(m0 + p) * Kdim + kb + kk);
        }
    };

    stage(0, 0);  // prologue: tile 0 -> buf 0
    int it = 0;
    for (int kb = 0; kb < Kdim; kb += 32, ++it) {
        int cur = it & 1;
        bool has_next = (kb + 32 < Kdim);
        if (has_next) stage(cur ^ 1, kb + 32);   // overlap next copy with this compute
        wait_async_le(has_next ? 1 : 0);          // own-wave in-order: cur tile complete
        __syncthreads();                          // all waves' cur tile visible
        v16h bf = frag_load(brow, kb, hf);
        for (int mt = 0; mt < 4; ++mt) {
            v16h af = frag_load(At[cur] + (mt * 16 + l15) * 32, 0, hf);
            acc[mt] = wmma_f16(af, bf, acc[mt]);
        }
        __syncthreads();                          // done reading cur; safe to overwrite next iter
    }

    float bi = bias[n];
    float scale = g[n] * rsqrtf(rv[n] + 1e-5f);
    float mr = rm[n], bbn = bb[n];
    for (int mt = 0; mt < 4; ++mt)
        for (int v = 0; v < 8; ++v) {
            int m = m0 + mt * 16 + v + 8 * hf;
            float x = (acc[mt][v] + bi - mr) * scale + bbn;
            out[(size_t)m * N + n] = (_Float16)elu_f(x);
        }
}

// ---------------------------------------------------------------- fc3 + sigmoid
__global__ void fc3_kernel(const _Float16* __restrict__ act4, const float* __restrict__ w,
                           const float* __restrict__ b, float* __restrict__ out, int Bn) {
    int i = blockIdx.x * blockDim.x + threadIdx.x;
    if (i >= Bn) return;
    float s = b[0];
    for (int j = 0; j < 64; ++j) s += (float)act4[(size_t)i * 64 + j] * w[j];
    out[i] = 1.f / (1.f + __expf(-s));
}

// ---------------------------------------------------------------- host launcher
extern "C" void kernel_launch(void* const* d_in, const int* in_sizes, int n_in,
                              void* d_out, int out_size, void* d_ws, size_t ws_size,
                              hipStream_t stream) {
    constexpr int B = 4096, E = 128;
    const int*   user    = (const int*)d_in[0];
    const int*   item    = (const int*)d_in[1];
    const float* eps_u   = (const float*)d_in[2];
    const float* eps_i   = (const float*)d_in[3];
    const float* Wum = (const float*)d_in[4],  *bum = (const float*)d_in[5];
    const float* Wuv = (const float*)d_in[6],  *buv = (const float*)d_in[7];
    const float* Wim = (const float*)d_in[8],  *bim = (const float*)d_in[9];
    const float* Wiv = (const float*)d_in[10], *biv = (const float*)d_in[11];
    const float* conv1_w = (const float*)d_in[12], *conv1_b = (const float*)d_in[13];
    const float* conv2_w = (const float*)d_in[14], *conv2_b = (const float*)d_in[15];
    const float* fc1_w = (const float*)d_in[16], *fc1_b = (const float*)d_in[17];
    const float* bn1_g = (const float*)d_in[18], *bn1_b = (const float*)d_in[19];
    const float* bn1_rm = (const float*)d_in[20], *bn1_rv = (const float*)d_in[21];
    const float* fc2_w = (const float*)d_in[22], *fc2_b = (const float*)d_in[23];
    const float* bn2_g = (const float*)d_in[24], *bn2_b = (const float*)d_in[25];
    const float* bn2_rm = (const float*)d_in[26], *bn2_rv = (const float*)d_in[27];
    const float* fc3_w = (const float*)d_in[28], *fc3_b = (const float*)d_in[29];

    // workspace layout (bytes, 256-aligned regions)
    char* ws = (char*)d_ws;
    constexpr size_t SZ_SUSI = (size_t)B * 9 * E * sizeof(_Float16);
    constexpr size_t OFF_SU   = 0;
    constexpr size_t OFF_SI   = OFF_SU + SZ_SUSI;
    constexpr size_t OFF_W1H  = OFF_SI + SZ_SUSI;                    // 64*128
    constexpr size_t OFF_W2H  = OFF_W1H + 64 * 128 * 2;              // 64*576
    constexpr size_t OFF_WF1H = OFF_W2H + 64 * 576 * 2;              // 512*3136
    constexpr size_t OFF_WF2H = OFF_WF1H + (size_t)512 * 3136 * 2;   // 64*512
    constexpr size_t OFF_A1   = OFF_WF2H + 64 * 512 * 2;             // B*64*81
    constexpr size_t OFF_A2   = OFF_A1 + (size_t)B * 64 * 81 * 2;    // B*3136
    constexpr size_t OFF_A3   = OFF_A2 + (size_t)B * 3136 * 2;       // B*512
    constexpr size_t OFF_A4   = OFF_A3 + (size_t)B * 512 * 2;        // B*64

    _Float16* su   = (_Float16*)(ws + OFF_SU);
    _Float16* si   = (_Float16*)(ws + OFF_SI);
    _Float16* w1h  = (_Float16*)(ws + OFF_W1H);
    _Float16* w2h  = (_Float16*)(ws + OFF_W2H);
    _Float16* wf1h = (_Float16*)(ws + OFF_WF1H);
    _Float16* wf2h = (_Float16*)(ws + OFF_WF2H);
    _Float16* act1 = (_Float16*)(ws + OFF_A1);
    _Float16* act2 = (_Float16*)(ws + OFF_A2);
    _Float16* act3 = (_Float16*)(ws + OFF_A3);
    _Float16* act4 = (_Float16*)(ws + OFF_A4);

    auto cvt = [&](const float* s, _Float16* d, int n) {
        cvt_f32_f16<<<(n + 255) / 256, 256, 0, stream>>>(s, d, n);
    };
    cvt(conv1_w, w1h, 64 * 128);
    cvt(conv2_w, w2h, 64 * 576);
    cvt(fc1_w, wf1h, 512 * 3136);
    cvt(fc2_w, wf2h, 64 * 512);

    embed_kernel<<<B, 128, 0, stream>>>(user, item, eps_u, eps_i,
                                        Wum, bum, Wuv, buv, Wim, bim, Wiv, biv, su, si);
    conv1_kernel<<<B, 128, 0, stream>>>(su, si, w1h, conv1_b, act1);
    conv2_kernel<<<B, 128, 0, stream>>>(act1, w2h, conv2_b, act2);
    fc_gemm_kernel<<<dim3(B / 64, 512 / 64), 128, 0, stream>>>(
        act2, wf1h, fc1_b, bn1_g, bn1_b, bn1_rm, bn1_rv, act3, 512, 3136);
    fc_gemm_kernel<<<dim3(B / 64, 1), 128, 0, stream>>>(
        act3, wf2h, fc2_b, bn2_g, bn2_b, bn2_rm, bn2_rv, act4, 64, 512);
    fc3_kernel<<<(B + 255) / 256, 256, 0, stream>>>(act4, fc3_w, fc3_b, (float*)d_out, B);
}
